// Decoder_59725815218578
// MI455X (gfx1250) — compile-verified, matrix operand order checked by reading, not run on hardware
//
#include <hip/hip_runtime.h>

// ---------------- constants ----------------
constexpr int cB = 8, cT = 512, cS = 512, cE = 512, cH = 1024, cV = 32000, cL = 6;
constexpr int TP = cT + 2;                 // padded time rows (K-1 = 2 left pad)
constexpr float SCALE = 0.70710678118f;    // sqrt(0.5)

typedef __attribute__((ext_vector_type(16))) __bf16        v16bf;
typedef __attribute__((ext_vector_type(8)))  float         v8f;
typedef __attribute__((ext_vector_type(8)))  unsigned int  v8u;

// ---------------- helpers ----------------
__device__ __forceinline__ unsigned short f2bf(float x) {
  unsigned int u = __builtin_bit_cast(unsigned int, x);
  unsigned int r = u + 0x7FFFu + ((u >> 16) & 1u);   // round-to-nearest-even
  return (unsigned short)(r >> 16);
}

__device__ __forceinline__ v8f wmma_bf16(v16bf a, v16bf b, v8f c) {
  return __builtin_amdgcn_wmma_f32_16x16x32_bf16(false, a, false, b, (short)0, c, false, false);
}

// Load a 16x32 bf16 fragment (A or B^T, row-major, k contiguous, LDS row stride
// 40 u16). Per CDNA5 ISA 7.12.2 the 16 bf16 a lane owns are two 16-byte runs:
// bytes [16*half, +16) and [32+16*half, +16) of row (lane&15). Two ds_load_b128.
__device__ __forceinline__ v16bf frag_ld(const unsigned short* p, int lane) {
  const int mr = lane & 15, half = lane >> 4;
  const unsigned short* q = p + mr * 40 + half * 8;
  const uint4 lo = *(const uint4*)(q);
  const uint4 hi = *(const uint4*)(q + 16);
  v8u u;
  u[0] = lo.x; u[1] = lo.y; u[2] = lo.z; u[3] = lo.w;
  u[4] = hi.x; u[5] = hi.y; u[6] = hi.z; u[7] = hi.w;
  return __builtin_bit_cast(v16bf, u);
}

// ---------------- GEMM: C[M,N] = A[M,K](bf16) * Bt[N,K](bf16), fused epilogue ----
// Block tile 128(M) x 128(N) x 32(K); 8 waves (2x4), wave tile 64x32 = 4x2 WMMA.
// Double-buffered LDS: one barrier per K-step; global loads for k+32 issued before
// compute so their latency hides under 8 WMMAs.
__global__ __launch_bounds__(256) void gemm_bf16_kernel(
    const unsigned short* __restrict__ A,  long long lda,  long long sA,
    const unsigned short* __restrict__ Bt, long long ldb,  long long sB,
    const float* __restrict__ bias,
    const float* __restrict__ res, long long ldr, long long sR,
    float scale,
    float* __restrict__ oF, long long ldcF, long long sF,
    unsigned short* __restrict__ oH, long long ldcH, long long sH,
    int M, int N, int Kd) {
  __shared__ __attribute__((aligned(16))) unsigned short As[2][128][40];
  __shared__ __attribute__((aligned(16))) unsigned short Bs[2][128][40];

  const int t = threadIdx.x;
  const int lane = t & 31, wave = t >> 5;
  const int wm = (wave >> 2) * 64, wn = (wave & 3) * 32;
  const long long m0 = (long long)blockIdx.y * 128;
  const long long n0 = (long long)blockIdx.x * 128;

  A  += (long long)blockIdx.z * sA;
  Bt += (long long)blockIdx.z * sB;
  if (res) res += (long long)blockIdx.z * sR;
  if (oF)  oF  += (long long)blockIdx.z * sF;
  if (oH)  oH  += (long long)blockIdx.z * sH;

  const int ar = t >> 2, ac = (t & 3) * 8;            // 256 thr: 64 rows x 32 cols
  const long long aoff0 = (m0 + ar)      * lda + ac;  // A rows 0..63
  const long long aoff1 = (m0 + ar + 64) * lda + ac;  // A rows 64..127
  const long long boff0 = (n0 + ar)      * ldb + ac;  // B rows 0..63
  const long long boff1 = (n0 + ar + 64) * ldb + ac;  // B rows 64..127

  // ---- prologue: fill buffer 0
  uint4 av0 = *(const uint4*)(A  + aoff0);
  uint4 av1 = *(const uint4*)(A  + aoff1);
  uint4 bv0 = *(const uint4*)(Bt + boff0);
  uint4 bv1 = *(const uint4*)(Bt + boff1);
  *(uint4*)&As[0][ar][ac]      = av0;
  *(uint4*)&As[0][ar + 64][ac] = av1;
  *(uint4*)&Bs[0][ar][ac]      = bv0;
  *(uint4*)&Bs[0][ar + 64][ac] = bv1;
  __syncthreads();

  v8f acc[4][2] = {};
  int p = 0;

  for (int k0 = 0; k0 < Kd; k0 += 32) {
    const bool more = (k0 + 32) < Kd;
    if (more) {
      const int kn = k0 + 32;
      av0 = *(const uint4*)(A  + aoff0 + kn);
      av1 = *(const uint4*)(A  + aoff1 + kn);
      bv0 = *(const uint4*)(Bt + boff0 + kn);
      bv1 = *(const uint4*)(Bt + boff1 + kn);
      if (kn + 32 < Kd) {                             // gfx1250 global_prefetch_b8
        __builtin_prefetch(A  + aoff0 + kn + 32, 0, 1);
        __builtin_prefetch(A  + aoff1 + kn + 32, 0, 1);
        __builtin_prefetch(Bt + boff0 + kn + 32, 0, 1);
        __builtin_prefetch(Bt + boff1 + kn + 32, 0, 1);
      }
    }

    // ---- compute from buffer p: 8 WMMAs from 6 fragment loads
    const v16bf bf0 = frag_ld(&Bs[p][wn][0],      lane);
    const v16bf bf1 = frag_ld(&Bs[p][wn + 16][0], lane);
#pragma unroll
    for (int i = 0; i < 4; ++i) {
      const v16bf af = frag_ld(&As[p][wm + i * 16][0], lane);
      acc[i][0] = wmma_bf16(af, bf0, acc[i][0]);
      acc[i][1] = wmma_bf16(af, bf1, acc[i][1]);
    }

    if (more) {
      *(uint4*)&As[1 - p][ar][ac]      = av0;
      *(uint4*)&As[1 - p][ar + 64][ac] = av1;
      *(uint4*)&Bs[1 - p][ar][ac]      = bv0;
      *(uint4*)&Bs[1 - p][ar + 64][ac] = bv1;
      __syncthreads();
      p ^= 1;
    }
  }

  // ---- epilogue: C/D layout — lane n = lane&15, vgpr r -> m = r + 8*(lane>>4)
  const int ln = lane & 15, half = lane >> 4;
#pragma unroll
  for (int i = 0; i < 4; ++i)
#pragma unroll
    for (int j = 0; j < 2; ++j) {
      const long long n = n0 + wn + j * 16 + ln;
      const float bv = bias ? bias[n] : 0.0f;
#pragma unroll
      for (int r = 0; r < 8; ++r) {
        const long long m = m0 + wm + i * 16 + half * 8 + r;
        float v = acc[i][j][r] + bv;
        if (res) v += res[m * ldr + n];
        v *= scale;
        if (oF) oF[m * ldcF + n] = v;
        if (oH) oH[m * ldcH + n] = f2bf(v);
      }
    }
}

// ---------------- elementwise / prep kernels ----------------
__global__ void to_bf16_kernel(const float* __restrict__ x, unsigned short* __restrict__ y,
                               long long n) {
  long long i = (long long)blockIdx.x * 256 + threadIdx.x;
  if (i < n) y[i] = f2bf(x[i]);
}

// y[c*R + r] = x[r*C + c]  (per batch slice via blockIdx.y)
__global__ void transpose_bf16_kernel(const float* __restrict__ x, unsigned short* __restrict__ y,
                                      int R, int C, long long sx, long long sy) {
  long long i = (long long)blockIdx.x * 256 + threadIdx.x;
  long long total = (long long)R * C;
  if (i >= total) return;
  int r = (int)(i / C), c = (int)(i % C);
  y[(long long)blockIdx.y * sy + (long long)c * R + r] =
      f2bf(x[(long long)blockIdx.y * sx + i]);
}

// conv_w (L,2H,H,3) -> wt (L, 2H, 3H) with wt[l][o][k*H+h]
__global__ void convw_bf16_kernel(const float* __restrict__ w, unsigned short* __restrict__ wt) {
  long long i = (long long)blockIdx.x * 256 + threadIdx.x;
  long long total = (long long)cL * 2 * cH * cH * 3;
  if (i >= total) return;
  int k = (int)(i % 3); long long r = i / 3;
  int h = (int)(r % cH); r /= cH;
  int o = (int)(r % (2 * cH)); int l = (int)(r / (2 * cH));
  wt[((long long)l * 2 * cH + o) * (3 * cH) + (long long)k * cH + h] = f2bf(w[i]);
}

__global__ void pad_init_kernel(float* __restrict__ pf, unsigned short* __restrict__ ph) {
  long long i = (long long)blockIdx.x * 256 + threadIdx.x;
  long long total = (long long)cB * 2 * cH;
  if (i >= total) return;
  int h = (int)(i % cH); int r = (int)((i / cH) % 2); int b = (int)(i / (2 * cH));
  long long o = ((long long)b * TP + r) * cH + h;
  pf[o] = 1.0f;            // reference pads with trg_pad_idx (=1) as a float
  ph[o] = 0x3F80;          // bf16(1.0)
}

__global__ void embed_kernel(const int* __restrict__ trg, const float* __restrict__ tok,
                             const float* __restrict__ pos,
                             float* __restrict__ ef, unsigned short* __restrict__ eh) {
  long long i = (long long)blockIdx.x * 256 + threadIdx.x;
  long long total = (long long)cB * cT * cE;
  if (i >= total) return;
  int e = (int)(i % cE); long long bt = i / cE; int tt = (int)(bt % cT);
  float v = tok[(long long)trg[bt] * cE + e] + pos[(long long)tt * cE + e];
  ef[i] = v; eh[i] = f2bf(v);
}

__global__ void glu_kernel(const float* __restrict__ g2, float* __restrict__ cf,
                           unsigned short* __restrict__ ch) {
  long long i = (long long)blockIdx.x * 256 + threadIdx.x;
  long long total = (long long)cB * cT * cH;
  if (i >= total) return;
  long long bt = i / cH; int h = (int)(i % cH);
  float a = g2[bt * 2 * cH + h];
  float g = g2[bt * 2 * cH + cH + h];
  float v = a * (1.0f / (1.0f + __expf(-g)));
  cf[i] = v; ch[i] = f2bf(v);
}

// conved=(conved+attnProj)*s ; conv_input=(conved+conv_input)*s  -> padded buffers
__global__ void residual2_kernel(const float* __restrict__ cf, const float* __restrict__ ap,
                                 float* __restrict__ pf, unsigned short* __restrict__ ph) {
  long long i = (long long)blockIdx.x * 256 + threadIdx.x;
  long long total = (long long)cB * cT * cH;
  if (i >= total) return;
  int h = (int)(i % cH); int tt = (int)((i / cH) % cT); int b = (int)(i / ((long long)cT * cH));
  long long po = ((long long)b * TP + tt + 2) * cH + h;
  float v1 = (cf[i] + ap[i]) * SCALE;
  float v2 = (v1 + pf[po]) * SCALE;
  pf[po] = v2; ph[po] = f2bf(v2);
}

__global__ __launch_bounds__(256) void softmax_bf16_kernel(const float* __restrict__ e,
                                                           unsigned short* __restrict__ a, int S) {
  __shared__ float red[256];
  const long long row = blockIdx.x;
  const float* er = e + row * (long long)S;
  const int tid = threadIdx.x;
  float m = -3.4e38f;
  for (int i = tid; i < S; i += 256) m = fmaxf(m, er[i]);
  red[tid] = m; __syncthreads();
  for (int s2 = 128; s2 > 0; s2 >>= 1) { if (tid < s2) red[tid] = fmaxf(red[tid], red[tid + s2]); __syncthreads(); }
  m = red[0]; __syncthreads();
  float sum = 0.f;
  for (int i = tid; i < S; i += 256) sum += __expf(er[i] - m);
  red[tid] = sum; __syncthreads();
  for (int s2 = 128; s2 > 0; s2 >>= 1) { if (tid < s2) red[tid] += red[tid + s2]; __syncthreads(); }
  const float inv = 1.0f / red[0];
  for (int i = tid; i < S; i += 256) a[row * S + i] = f2bf(__expf(er[i] - m) * inv);
}

// ---------------- host side ----------------
static inline void launch_gemm(hipStream_t s,
    const unsigned short* A, long long lda, long long sA,
    const unsigned short* Bt, long long ldb, long long sB,
    const float* bias,
    const float* res, long long ldr, long long sR, float scale,
    float* oF, long long ldcF, long long sF,
    unsigned short* oH, long long ldcH, long long sH,
    int M, int N, int Kd, int batch) {
  dim3 g(N / 128, M / 128, batch), b(256);
  gemm_bf16_kernel<<<g, b, 0, s>>>(A, lda, sA, Bt, ldb, sB, bias, res, ldr, sR, scale,
                                   oF, ldcF, sF, oH, ldcH, sH, M, N, Kd);
}

extern "C" void kernel_launch(void* const* d_in, const int* in_sizes, int n_in,
                              void* d_out, int out_size, void* d_ws, size_t ws_size,
                              hipStream_t stream) {
  (void)in_sizes; (void)n_in; (void)out_size; (void)ws_size;
  const int*   trg     = (const int*)  d_in[0];
  const float* encCv   = (const float*)d_in[1];
  const float* encCb   = (const float*)d_in[2];
  const float* tokEmb  = (const float*)d_in[3];
  const float* posEmb  = (const float*)d_in[4];
  const float* e2h_w   = (const float*)d_in[5];
  const float* e2h_b   = (const float*)d_in[6];
  const float* h2e_w   = (const float*)d_in[7];
  const float* h2e_b   = (const float*)d_in[8];
  const float* ah2e_w  = (const float*)d_in[9];
  const float* ah2e_b  = (const float*)d_in[10];
  const float* ae2h_w  = (const float*)d_in[11];
  const float* ae2h_b  = (const float*)d_in[12];
  const float* fc_w    = (const float*)d_in[13];
  const float* fc_b    = (const float*)d_in[14];
  const float* conv_w  = (const float*)d_in[15];
  const float* conv_b  = (const float*)d_in[16];

  char* ws = (char*)d_ws;
  size_t off = 0;
  auto take = [&](size_t bytes) -> char* {
    char* p = ws + off; off += (bytes + 255) & ~(size_t)255; return p;
  };
  const long long BT = (long long)cB * cT;

  float*          embF     = (float*)         take(BT * cE * 4);
  unsigned short* embH     = (unsigned short*)take(BT * cE * 2);
  unsigned short* e2hWT    = (unsigned short*)take((long long)cH * cE * 2);
  unsigned short* h2eWT    = (unsigned short*)take((long long)cE * cH * 2);
  unsigned short* ah2eWT   = (unsigned short*)take((long long)cE * cH * 2);
  unsigned short* ae2hWT   = (unsigned short*)take((long long)cH * cE * 2);
  unsigned short* fcWT     = (unsigned short*)take((long long)cV * cE * 2);
  unsigned short* convWT   = (unsigned short*)take((long long)cL * 2 * cH * 3 * cH * 2);
  unsigned short* encCvH   = (unsigned short*)take((long long)cB * cS * cE * 2);
  unsigned short* encCbT   = (unsigned short*)take((long long)cB * cE * cS * 2);
  float*          convPadF = (float*)         take((long long)cB * TP * cH * 4);
  unsigned short* convPadH = (unsigned short*)take((long long)cB * TP * cH * 2);
  float*          gluF     = (float*)         take(BT * 2 * cH * 4);
  float*          convedF  = (float*)         take(BT * cH * 4);
  unsigned short* convedH  = (unsigned short*)take(BT * cH * 2);
  unsigned short* combH    = (unsigned short*)take(BT * cE * 2);
  float*          energyF  = (float*)         take(BT * cS * 4);
  unsigned short* attnH    = (unsigned short*)take(BT * cS * 2);
  unsigned short* attendH  = (unsigned short*)take(BT * cE * 2);
  float*          aprojF   = (float*)         take(BT * cH * 4);
  unsigned short* convOutH = (unsigned short*)take(BT * cE * 2);

  auto blocks = [](long long n) { return (unsigned)((n + 255) / 256); };

  // ---- prep: bf16 weights (N-major, k-contiguous), encoder copies, pads, embeddings
  transpose_bf16_kernel<<<dim3(blocks((long long)cE * cH), 1), 256, 0, stream>>>(e2h_w,  e2hWT,  cE, cH, 0, 0);
  transpose_bf16_kernel<<<dim3(blocks((long long)cH * cE), 1), 256, 0, stream>>>(h2e_w,  h2eWT,  cH, cE, 0, 0);
  transpose_bf16_kernel<<<dim3(blocks((long long)cH * cE), 1), 256, 0, stream>>>(ah2e_w, ah2eWT, cH, cE, 0, 0);
  transpose_bf16_kernel<<<dim3(blocks((long long)cE * cH), 1), 256, 0, stream>>>(ae2h_w, ae2hWT, cE, cH, 0, 0);
  transpose_bf16_kernel<<<dim3(blocks((long long)cE * cV), 1), 256, 0, stream>>>(fc_w,   fcWT,   cE, cV, 0, 0);
  transpose_bf16_kernel<<<dim3(blocks((long long)cS * cE), cB), 256, 0, stream>>>(encCb, encCbT, cS, cE,
      (long long)cS * cE, (long long)cE * cS);
  to_bf16_kernel<<<blocks((long long)cB * cS * cE), 256, 0, stream>>>(encCv, encCvH, (long long)cB * cS * cE);
  convw_bf16_kernel<<<blocks((long long)cL * 2 * cH * cH * 3), 256, 0, stream>>>(conv_w, convWT);
  pad_init_kernel<<<blocks((long long)cB * 2 * cH), 256, 0, stream>>>(convPadF, convPadH);
  embed_kernel<<<blocks(BT * cE), 256, 0, stream>>>(trg, tokEmb, posEmb, embF, embH);

  // ---- emb2hid: (T,E)@(E,H)+b -> padded conv input (f32 + bf16), batched over B
  launch_gemm(stream, embH, cE, (long long)cT * cE, e2hWT, cE, 0, e2h_b,
              nullptr, 0, 0, 1.0f,
              convPadF + 2 * cH, cH, (long long)TP * cH,
              convPadH + 2 * cH, cH, (long long)TP * cH,
              cT, cH, cE, cB);

  for (int l = 0; l < cL; ++l) {
    // conv as overlapping-row GEMM: A row t = padded rows t..t+2 (3H contiguous)
    launch_gemm(stream, convPadH, cH, (long long)TP * cH,
                convWT + (long long)l * 2 * cH * 3 * cH, 3 * cH, 0,
                conv_b + (long long)l * 2 * cH,
                nullptr, 0, 0, 1.0f,
                gluF, 2 * cH, (long long)cT * 2 * cH, nullptr, 0, 0,
                cT, 2 * cH, 3 * cH, cB);
    glu_kernel<<<blocks(BT * cH), 256, 0, stream>>>(gluF, convedF, convedH);

    // combined = (conved@W + b + embedded) * sqrt(.5)  (bf16 only, feeds energy)
    launch_gemm(stream, convedH, cH, 0, ah2eWT, cH, 0, ah2e_b,
                embF, cE, 0, SCALE,
                nullptr, 0, 0, combH, cE, 0,
                cB * cT, cE, cH, 1);

    // energy[b] = combined[b] @ enc_conved[b]^T   (enc is already k-contiguous)
    launch_gemm(stream, combH, cE, (long long)cT * cE, encCvH, cE, (long long)cS * cE,
                nullptr, nullptr, 0, 0, 1.0f,
                energyF, cS, (long long)cT * cS, nullptr, 0, 0,
                cT, cS, cE, cB);
    softmax_bf16_kernel<<<(unsigned)BT, 256, 0, stream>>>(energyF, attnH, cS);

    // attended[b] = attn[b] @ enc_combined[b]  (via enc_combined^T, k-contiguous)
    launch_gemm(stream, attnH, cS, (long long)cT * cS, encCbT, cS, (long long)cE * cS,
                nullptr, nullptr, 0, 0, 1.0f,
                nullptr, 0, 0, attendH, cE, (long long)cT * cE,
                cT, cE, cS, cB);

    // attnProj = attended @ W + b
    launch_gemm(stream, attendH, cE, 0, ae2hWT, cE, 0, ae2h_b,
                nullptr, 0, 0, 1.0f,
                aprojF, cH, 0, nullptr, 0, 0,
                cB * cT, cH, cE, 1);

    residual2_kernel<<<blocks(BT * cH), 256, 0, stream>>>(convedF, aprojF, convPadF, convPadH);
  }

  // hid2emb -> bf16 conv_output
  launch_gemm(stream, convPadH + 2 * cH, cH, (long long)TP * cH, h2eWT, cH, 0, h2e_b,
              nullptr, 0, 0, 1.0f,
              nullptr, 0, 0, convOutH, cE, (long long)cT * cE,
              cT, cE, cH, cB);

  // fc_out: (B*T, E) @ (E, V) + b -> logits (fp32 output)
  launch_gemm(stream, convOutH, cE, 0, fcWT, cE, 0, fc_b,
              nullptr, 0, 0, 1.0f,
              (float*)d_out, cV, 0, nullptr, 0, 0,
              cB * cT, cV, cE, 1);
}
